// DeepWalk_17214228922649
// MI455X (gfx1250) — matrix operand order repaired and checked by reading, not compiled, not run on hardware
//
#include <hip/hip_runtime.h>
#include <hip/hip_bf16.h>

// ---------------------------------------------------------------------------
// DeepWalk skip-gram NEG-sampling on MI455X (gfx1250, wave32).
//
// Roofline: ~655 MB of random 512B row gathers for the loss + ~128 MB for the
// embedding gathers vs ~0.3 GFLOP -> pure HBM-gather bound (~35us @ 23.3TB/s).
// Strategy: B128 coalesced loads everywhere; the 9 per-pair 128-dim dots are
// run on the matrix pipe: 16 pairs/wave, v_wmma_f32_16x16x32_f16 chained 4x
// over K=128 per direction (diag(A x B_j) = logits), f32 accumulation.
// A/B are loaded straight from global in a contiguous-K order (k = 16g+s), same for A and B
// (contiguous 32-64B runs per lane), f32->f16 converted in registers.
// ---------------------------------------------------------------------------

typedef __attribute__((ext_vector_type(16))) _Float16 v16h;
typedef __attribute__((ext_vector_type(8)))  float    v8f;

#define DIM        128
#define B_PAIRS    131072
#define NUM_NEGS   8
#define B_IN       65536
#define EMB_ELEMS  (B_IN * DIM)          // 8388608 floats per embedding output
#define LOSS_IDX   (2 * EMB_ELEMS)       // 16777216
#define MRR_IDX    (2 * EMB_ELEMS + 1)   // 16777217

// ------------------------------- helpers -----------------------------------

__device__ __forceinline__ v16h pack16(float4 r0, float4 r1, float4 r2, float4 r3) {
    v16h h;
    h[0]  = (_Float16)r0.x; h[1]  = (_Float16)r0.y; h[2]  = (_Float16)r0.z; h[3]  = (_Float16)r0.w;
    h[4]  = (_Float16)r1.x; h[5]  = (_Float16)r1.y; h[6]  = (_Float16)r1.z; h[7]  = (_Float16)r1.w;
    h[8]  = (_Float16)r2.x; h[9]  = (_Float16)r2.y; h[10] = (_Float16)r2.z; h[11] = (_Float16)r2.w;
    h[12] = (_Float16)r3.x; h[13] = (_Float16)r3.y; h[14] = (_Float16)r3.z; h[15] = (_Float16)r3.w;
    return h;
}

// select acc[r] with r in [0,8) -> v_cndmask chain, no memory
__device__ __forceinline__ float sel8(v8f a, int r) {
    float d = a[0];
    d = (r == 1) ? a[1] : d;
    d = (r == 2) ? a[2] : d;
    d = (r == 3) ? a[3] : d;
    d = (r == 4) ? a[4] : d;
    d = (r == 5) ? a[5] : d;
    d = (r == 6) ? a[6] : d;
    d = (r == 7) ? a[7] : d;
    return d;
}

// numerically safe softplus(x) = log1p(exp(x))
__device__ __forceinline__ float softplusf(float x) {
    return fmaxf(x, 0.0f) + log1pf(__expf(-fabsf(x)));
}

// ------------------------- scalar init (poisoned d_out) --------------------

__global__ void dw_init_kernel(float* __restrict__ out) {
    out[LOSS_IDX] = 0.0f;
    out[MRR_IDX]  = 0.0f;
}

// ---------------------- embedding gather: 1 wave = 1 row -------------------
// lane q loads float4 q of row inputs[row] from both tables -> 512B/row/table,
// fully coalesced global_load_b128 / global_store_b128.

__global__ __launch_bounds__(256) void dw_gather_kernel(
    const int*   __restrict__ inputs,
    const float* __restrict__ tgt,
    const float* __restrict__ ctx,
    float*       __restrict__ out)
{
    int tid = blockIdx.x * 256 + threadIdx.x;   // [0, B_IN*32)
    int row = tid >> 5;
    int q   = tid & 31;
    int idx = inputs[row];

    const float4* t4 = (const float4*)tgt;
    const float4* c4 = (const float4*)ctx;
    float4*       o4 = (float4*)out;

    float4 a = t4[(size_t)idx * 32 + q];
    float4 b = c4[(size_t)idx * 32 + q];
    o4[tid]                    = a;              // embedding
    o4[(EMB_ELEMS / 4) + tid]  = b;              // context_embedding
}

// ------------------- loss + mrr via WMMA diagonal trick --------------------
// Wave handles 16 pairs. A (16x128) rows = target[src]; for each of the 9
// directions j (pos, neg0..7), B cols = context rows; logit_m = diag(A x B)_m.
// A-layout (16b A 16x32, contiguous K, not ISA table order): lane L (m=L%16, g=L/16) holds
//   halves 0..15 -> K = kc*32 + 16g + h        (16 consecutive f32 = 4 float4)
// B-layout (16b B 32x16): lane L (n=L%16, g=L/16) holds K = kc*32 + 16g + h
//   (16 consecutive f32 = 4 float4).
// D-layout (f32 16x16): lanes 0..7 hold diag for pairs 0..7 (reg r = lane),
// lanes 24..31 hold diag for pairs 8..15 (reg r = lane & 7).

__global__ __launch_bounds__(256) void dw_loss_kernel(
    const int*   __restrict__ src,
    const int*   __restrict__ pos,
    const int*   __restrict__ negs,
    const float* __restrict__ tgt,
    const float* __restrict__ ctx,
    float*       __restrict__ out)
{
    const int lane = threadIdx.x & 31;
    const int wave = threadIdx.x >> 5;
    const int pairBase = blockIdx.x * 128 + wave * 16;   // 8 waves x 16 pairs
    const int g = lane >> 4;
    const int n = lane & 15;
    const int p = pairBase + n;                          // pair for column n

    // gather indices (tiny scattered int loads)
    const int srcIdx = src[p];
    int cIdx[9];
    cIdx[0] = pos[p];
#pragma unroll
    for (int j = 0; j < NUM_NEGS; ++j) cIdx[1 + j] = negs[p * NUM_NEGS + j];

    // ---- load A (4 K-chunks of 32) once, in WMMA layout ----
    const float4* t4 = (const float4*)tgt + (size_t)srcIdx * 32;
    v16h A0, A1, A2, A3;
    {
        int c1, c2;
        c1 = 0 * 8 + 4 * g; c2 = c1 + 2;
        A0 = pack16(t4[c1], t4[c1 + 1], t4[c2], t4[c2 + 1]);
        c1 = 1 * 8 + 4 * g; c2 = c1 + 2;
        A1 = pack16(t4[c1], t4[c1 + 1], t4[c2], t4[c2 + 1]);
        c1 = 2 * 8 + 4 * g; c2 = c1 + 2;
        A2 = pack16(t4[c1], t4[c1 + 1], t4[c2], t4[c2 + 1]);
        c1 = 3 * 8 + 4 * g; c2 = c1 + 2;
        A3 = pack16(t4[c1], t4[c1 + 1], t4[c2], t4[c2 + 1]);
    }

    const int  myR   = lane & 7;
    const bool valid = ((g == 0) && (n < 8)) || ((g == 1) && (n >= 8));

    float logit_pos = 0.0f;
    float lossAcc   = 0.0f;
    int   worse     = 0;

#pragma unroll
    for (int j = 0; j < 9; ++j) {
        const float4* c4 = (const float4*)ctx + (size_t)cIdx[j] * 32;
        v8f acc = {};
        {   // kc = 0..3, B chunk = 4 consecutive float4 starting at kc*8+4g
            int cb;
            cb = 0 * 8 + 4 * g;
            acc = __builtin_amdgcn_wmma_f32_16x16x32_f16(
                false, A0, false, pack16(c4[cb], c4[cb + 1], c4[cb + 2], c4[cb + 3]),
                (short)0, acc, false, false);
            cb = 1 * 8 + 4 * g;
            acc = __builtin_amdgcn_wmma_f32_16x16x32_f16(
                false, A1, false, pack16(c4[cb], c4[cb + 1], c4[cb + 2], c4[cb + 3]),
                (short)0, acc, false, false);
            cb = 2 * 8 + 4 * g;
            acc = __builtin_amdgcn_wmma_f32_16x16x32_f16(
                false, A2, false, pack16(c4[cb], c4[cb + 1], c4[cb + 2], c4[cb + 3]),
                (short)0, acc, false, false);
            cb = 3 * 8 + 4 * g;
            acc = __builtin_amdgcn_wmma_f32_16x16x32_f16(
                false, A3, false, pack16(c4[cb], c4[cb + 1], c4[cb + 2], c4[cb + 3]),
                (short)0, acc, false, false);
        }
        float d = sel8(acc, myR);       // diagonal element for this lane's pair
        if (j == 0) {
            logit_pos = d;
        } else {
            lossAcc += softplusf(d);                // labels=0 -> softplus(x)
            worse   += (d >= logit_pos) ? 1 : 0;    // ties toward negatives
        }
    }
    lossAcc += softplusf(-logit_pos);               // labels=1 -> softplus(-x)
    float mrrAcc = (1.0f / (float)(1 + worse)) * (1.0f / (float)B_PAIRS);

    if (!valid) { lossAcc = 0.0f; mrrAcc = 0.0f; }

    // wave32 butterfly reduction
#pragma unroll
    for (int m = 16; m >= 1; m >>= 1) {
        lossAcc += __shfl_xor(lossAcc, m, 32);
        mrrAcc  += __shfl_xor(mrrAcc,  m, 32);
    }
    if (lane == 0) {
        atomicAdd(&out[LOSS_IDX], lossAcc);
        atomicAdd(&out[MRR_IDX],  mrrAcc);
    }
}

// ------------------------------- launcher ----------------------------------

extern "C" void kernel_launch(void* const* d_in, const int* in_sizes, int n_in,
                              void* d_out, int out_size, void* d_ws, size_t ws_size,
                              hipStream_t stream) {
    const int*   inputs = (const int*)  d_in[0];   // [65536]
    const int*   src    = (const int*)  d_in[1];   // [131072,1]
    const int*   pos    = (const int*)  d_in[2];   // [131072,1]
    const int*   negs   = (const int*)  d_in[3];   // [131072,8]
    const float* tgt    = (const float*)d_in[4];   // [1e6,128]
    const float* ctx    = (const float*)d_in[5];   // [1e6,128]
    float*       out    = (float*)d_out;

    dw_init_kernel<<<1, 1, 0, stream>>>(out);
    dw_gather_kernel<<<(B_IN * 32) / 256, 256, 0, stream>>>(inputs, tgt, ctx, out);
    dw_loss_kernel<<<B_PAIRS / 128, 256, 0, stream>>>(src, pos, negs, tgt, ctx, out);
}
